// MultiHeadAttention_31413390802960
// MI455X (gfx1250) — compile-verified, matrix operand order checked by reading, not run on hardware
//
#include <hip/hip_runtime.h>

// ---------------------------------------------------------------------------
// MI455X (gfx1250) multi-head attention forward.
// B=4, L=2048, D=1024, H=16, DK=64.  f16 WMMA (16x16x32) with f32 accum.
// Pipeline: cvt(x) + transpose-cvt(W) -> QKV GEMM -> flash attention -> out GEMM
// Data movement uses gfx1250 GLOBAL_LOAD_ASYNC_TO_LDS_B128 (ASYNCcnt) for
// GEMM tiles and attention K blocks; causal mask handled in a split loop so
// the steady-state KV march carries no mask VALU.
// ---------------------------------------------------------------------------

typedef __attribute__((ext_vector_type(16))) _Float16     v16h;
typedef __attribute__((ext_vector_type(8)))  float        v8f;
typedef __attribute__((ext_vector_type(8)))  unsigned int v8u;

#define BATCH   4
#define LQ      2048
#define DMODEL  1024
#define NH      16
#define DKH     64

// Load one 16x32 f16 A/B fragment (v16h) from a 16B-aligned pointer.
// Lane (m=lane%16, half=lane/16) holds K = half*8 + {0..7} in VGPR0..3 and
// K = 16 + half*8 + {0..7} in VGPR4..7 -> two b128 loads.
static __device__ inline v16h load_frag(const _Float16* p) {
  v8u u;
  uint4 a = *(const uint4*)(p);
  uint4 b = *(const uint4*)(p + 16);
  u[0] = a.x; u[1] = a.y; u[2] = a.z; u[3] = a.w;
  u[4] = b.x; u[5] = b.y; u[6] = b.z; u[7] = b.w;
  return __builtin_bit_cast(v16h, u);
}

static __device__ inline v8f wmma_f16(v16h a, v16h b, v8f c) {
  return __builtin_amdgcn_wmma_f32_16x16x32_f16(
      /*neg_a=*/false, a, /*neg_b=*/false, b,
      /*c_mod=*/(short)0, c, /*reuse_a=*/false, /*reuse_b=*/false);
}

// gfx1250 async memory->LDS copy (16B per lane), tracked by ASYNCcnt.
// vdst = per-lane LDS byte offset (generic LDS addr low 32 bits),
// vaddr = 64-bit global VA, saddr = off.
static __device__ inline void async_load_b128(const _Float16* gp, _Float16* lp) {
  unsigned lds_off = (unsigned)(size_t)lp;          // low 32 bits = LDS offset
  unsigned long long ga = (unsigned long long)(size_t)gp;
  asm volatile("global_load_async_to_lds_b128 %0, %1, off"
               :: "v"(lds_off), "v"(ga)
               : "memory");
}

#define WAIT_ASYNCCNT(n) asm volatile("s_wait_asynccnt " #n ::: "memory")

// ---------------------------------------------------------------------------
// Prep kernels.
// ---------------------------------------------------------------------------
__global__ void cvt_f32_f16(const float* __restrict__ s,
                            _Float16* __restrict__ d, int n) {
  int i = blockIdx.x * blockDim.x + threadIdx.x;
  if (i < n) d[i] = (_Float16)s[i];
}

__global__ void transpose_cvt(const float* __restrict__ src,
                              _Float16* __restrict__ dst,
                              int rows, int cols) {
  int c = blockIdx.x * 32 + (threadIdx.x & 31);
  int r = blockIdx.y * 8  + (threadIdx.x >> 5);
  if (r < rows && c < cols)
    dst[(size_t)c * rows + r] = (_Float16)src[(size_t)r * cols + c];
}

// ---------------------------------------------------------------------------
// Tiled WMMA GEMM: C[M,N] = A[M,1024] * B[1024,N] + bias, A/Bt f16.
// BLOCK 128x128x64, 256 threads = 8 waves (2M x 4N), each wave 4x2 WMMA accums.
// Double-buffered LDS filled by GLOBAL_LOAD_ASYNC_TO_LDS_B128:
//   stage(buf^1, k0+64); s_wait_asynccnt 8; barrier; compute(buf).
// MODE 0: QKV epilogue -> scatter Q*0.125 / K / V (f16, [B,H,L,DK]).
// MODE 1: out epilogue -> f32 C.
// ---------------------------------------------------------------------------
template <int MODE>
__global__ __launch_bounds__(256) void gemm128(
    const _Float16* __restrict__ A, const _Float16* __restrict__ Bt,
    const float* __restrict__ bias,
    _Float16* __restrict__ qo, _Float16* __restrict__ ko,
    _Float16* __restrict__ vo, float* __restrict__ co) {
  __shared__ __align__(16) _Float16 As[2][128 * 72];  // 64 K + pad(8)
  __shared__ __align__(16) _Float16 Bs[2][128 * 72];

  const int m0 = blockIdx.y * 128;
  const int n0 = blockIdx.x * 128;
  const int t = threadIdx.x;
  const int wave = t >> 5, lane = t & 31;
  const int hf = lane >> 4, ln = lane & 15;
  const int wm = wave >> 2, wn = wave & 3;

  v8f zero = {0.f, 0.f, 0.f, 0.f, 0.f, 0.f, 0.f, 0.f};
  v8f acc[4][2];
#pragma unroll
  for (int i = 0; i < 4; i++)
#pragma unroll
    for (int j = 0; j < 2; j++) acc[i][j] = zero;

  const int lrow = t >> 1;          // 0..127
  const int lcol = (t & 1) * 32;    // 0 or 32

  // Issue one 128x64 tile of A and Bt: 4 + 4 async b128 per thread
  // (= 8 ASYNCcnt increments per wave per stage).
  auto stage = [&](int buf, int k0) {
    const _Float16* ap = A + (size_t)(m0 + lrow) * DMODEL + k0 + lcol;
    const _Float16* bp = Bt + (size_t)(n0 + lrow) * DMODEL + k0 + lcol;
    _Float16* la = &As[buf][lrow * 72 + lcol];
    _Float16* lb = &Bs[buf][lrow * 72 + lcol];
#pragma unroll
    for (int i = 0; i < 4; i++) async_load_b128(ap + i * 8, la + i * 8);
#pragma unroll
    for (int i = 0; i < 4; i++) async_load_b128(bp + i * 8, lb + i * 8);
  };

  stage(0, 0);
  int buf = 0;
  for (int k0 = 0; k0 < DMODEL; k0 += 64) {
    if (k0 + 64 < DMODEL) {
      stage(buf ^ 1, k0 + 64);      // 8 new in flight on top of 8 old
      WAIT_ASYNCCNT(8);             // previous tile fully landed in LDS
    } else {
      WAIT_ASYNCCNT(0);
    }
    __syncthreads();                // everyone's slice of this tile complete

#pragma unroll
    for (int ks = 0; ks < 2; ks++) {
      v16h a[4], b[2];
#pragma unroll
      for (int mt = 0; mt < 4; mt++)
        a[mt] = load_frag(&As[buf][(wm * 64 + mt * 16 + ln) * 72 + ks * 32 + hf * 8]);
#pragma unroll
      for (int nt = 0; nt < 2; nt++)
        b[nt] = load_frag(&Bs[buf][(wn * 32 + nt * 16 + ln) * 72 + ks * 32 + hf * 8]);
#pragma unroll
      for (int mt = 0; mt < 4; mt++)
#pragma unroll
        for (int nt = 0; nt < 2; nt++)
          acc[mt][nt] = wmma_f16(a[mt], b[nt], acc[mt][nt]);
    }
    __syncthreads();                // reads done before buf is refilled next iter
    buf ^= 1;
  }

#pragma unroll
  for (int mt = 0; mt < 4; mt++) {
#pragma unroll
    for (int nt = 0; nt < 2; nt++) {
#pragma unroll
      for (int r = 0; r < 8; r++) {
        int row = m0 + wm * 64 + mt * 16 + r + 8 * hf;  // C: M = r + 8*half
        int col = n0 + wn * 32 + nt * 16 + ln;          // C: N = lane%16
        float val = acc[mt][nt][r] + bias[col];
        if (MODE == 0) {
          int bb = row >> 11, ll = row & (LQ - 1);
          int part = col >> 10, cc = col & (DMODEL - 1);
          int hh = cc >> 6, dk = cc & (DKH - 1);
          size_t idx = (((size_t)(bb * NH + hh)) * LQ + ll) * DKH + dk;
          if (part == 0)       qo[idx] = (_Float16)(val * 0.125f);  // 1/sqrt(64)
          else if (part == 1)  ko[idx] = (_Float16)val;
          else                 vo[idx] = (_Float16)val;
        } else {
          co[(size_t)row * DMODEL + col] = val;
        }
      }
    }
  }
}

// ---------------------------------------------------------------------------
// Flash attention, causal.  grid = (L/128, B*H), 256 threads (8 waves).
// Each wave owns 16 query rows; KV marched in 64-key blocks.
// K block staged ONCE per workgroup into LDS via async loads (was 8x
// duplicated global traffic); V^T staged with an in-register transpose.
// Mask-free main loop + 2-iteration masked diagonal tail.
// ---------------------------------------------------------------------------
__global__ __launch_bounds__(256) void flash_attn(
    const _Float16* __restrict__ Q, const _Float16* __restrict__ K,
    const _Float16* __restrict__ V, _Float16* __restrict__ O) {
  __shared__ __align__(16) _Float16 Ps[8][16][72];  // per-wave P tile
  __shared__ __align__(16) _Float16 Ks[64 * 72];    // K block (key-major)
  __shared__ __align__(16) _Float16 Vt[64][72];     // V^T (dk-major)

  const int bh = blockIdx.y;
  const int b = bh >> 4, h = bh & (NH - 1);
  const int q0 = blockIdx.x * 128;
  const int t = threadIdx.x;
  const int wave = t >> 5, lane = t & 31;
  const int hf = lane >> 4, ln = lane & 15;
  const size_t base = (size_t)bh * LQ * DKH;

  // Q fragments for this wave's 16 rows, resident for the whole kernel.
  v16h aq[2];
  {
    int row = q0 + wave * 16 + ln;
    const _Float16* qp = Q + base + (size_t)row * DKH + hf * 8;
    aq[0] = load_frag(qp);        // dk 0..31
    aq[1] = load_frag(qp + 32);   // dk 32..63
  }

  v8f zero = {0.f, 0.f, 0.f, 0.f, 0.f, 0.f, 0.f, 0.f};
  float mrow[8], lsum[8];
  v8f oacc[4];
#pragma unroll
  for (int r = 0; r < 8; r++) { mrow[r] = -3.0e38f; lsum[r] = 0.f; }
#pragma unroll
  for (int d = 0; d < 4; d++) oacc[d] = zero;

  // One 64-key block; `masked` is a call-site constant, folded after inlining.
  auto process_block = [&](int j0, bool masked) {
    __syncthreads();  // previous iteration's Ks/Vt readers are done

    // ---- async-stage K block [64 keys][64 dk] into LDS (2 ops/thread) ----
    {
      int row = t >> 2;               // key 0..63
      int col = (t & 3) * 16;         // dk {0,16,32,48}
      const _Float16* kp = K + base + (size_t)(j0 + row) * DKH + col;
      _Float16* lk = &Ks[row * 72 + col];
      async_load_b128(kp, lk);
      async_load_b128(kp + 8, lk + 8);
    }
    // ---- stage V^T block (register transpose) ----
    {
      int key = t >> 2;
      int dk0 = (t & 3) * 16;
      const _Float16* vp = V + base + (size_t)(j0 + key) * DKH + dk0;
      uint4 u0 = *(const uint4*)vp;
      uint4 u1 = *(const uint4*)(vp + 8);
      const _Float16* h0 = (const _Float16*)&u0;
      const _Float16* h1 = (const _Float16*)&u1;
#pragma unroll
      for (int i = 0; i < 8; i++) Vt[dk0 + i][key]     = h0[i];
#pragma unroll
      for (int i = 0; i < 8; i++) Vt[dk0 + 8 + i][key] = h1[i];
    }
    WAIT_ASYNCCNT(0);   // K block landed in LDS (ds stores waited by compiler)
    __syncthreads();

    // ---- S = Q K^T  (16 x 64 per wave) ----
    v8f sacc[4];
#pragma unroll
    for (int nt = 0; nt < 4; nt++) sacc[nt] = zero;
#pragma unroll
    for (int nt = 0; nt < 4; nt++) {
      const _Float16* kp = &Ks[(nt * 16 + ln) * 72 + hf * 8];
      v16h bk0 = load_frag(kp);
      v16h bk1 = load_frag(kp + 32);
      sacc[nt] = wmma_f16(aq[0], bk0, sacc[nt]);
      sacc[nt] = wmma_f16(aq[1], bk1, sacc[nt]);
    }

    // ---- (mask) + online softmax (rows live in one 16-lane half) ----
#pragma unroll
    for (int r = 0; r < 8; r++) {
      float mx = -3.0e38f;
      if (masked) {
        int qi = q0 + wave * 16 + r + 8 * hf;
#pragma unroll
        for (int nt = 0; nt < 4; nt++) {
          int kj = j0 + nt * 16 + ln;
          float s = (kj <= qi) ? sacc[nt][r] : -1.0e30f;
          sacc[nt][r] = s;
          mx = fmaxf(mx, s);
        }
      } else {
#pragma unroll
        for (int nt = 0; nt < 4; nt++) mx = fmaxf(mx, sacc[nt][r]);
      }
      mx = fmaxf(mx, __shfl_xor(mx, 1, 32));
      mx = fmaxf(mx, __shfl_xor(mx, 2, 32));
      mx = fmaxf(mx, __shfl_xor(mx, 4, 32));
      mx = fmaxf(mx, __shfl_xor(mx, 8, 32));
      float mnew = fmaxf(mrow[r], mx);
      float scale = __expf(mrow[r] - mnew);
      float rs = 0.f;
#pragma unroll
      for (int nt = 0; nt < 4; nt++) {
        float p = __expf(sacc[nt][r] - mnew);
        Ps[wave][r + 8 * hf][nt * 16 + ln] = (_Float16)p;  // C->A via LDS
        rs += p;
      }
      rs += __shfl_xor(rs, 1, 32);
      rs += __shfl_xor(rs, 2, 32);
      rs += __shfl_xor(rs, 4, 32);
      rs += __shfl_xor(rs, 8, 32);
      lsum[r] = lsum[r] * scale + rs;
      mrow[r] = mnew;
#pragma unroll
      for (int d = 0; d < 4; d++) oacc[d][r] *= scale;
    }
    __syncthreads();

    // ---- O += P V ----
    v16h ap0 = load_frag(&Ps[wave][ln][hf * 8]);        // keys 0..31
    v16h ap1 = load_frag(&Ps[wave][ln][32 + hf * 8]);   // keys 32..63
#pragma unroll
    for (int d = 0; d < 4; d++) {
      v16h bv0 = load_frag(&Vt[d * 16 + ln][hf * 8]);
      v16h bv1 = load_frag(&Vt[d * 16 + ln][32 + hf * 8]);
      oacc[d] = wmma_f16(ap0, bv0, oacc[d]);
      oacc[d] = wmma_f16(ap1, bv1, oacc[d]);
    }
  };

  // Mask-free main march, then the 2-block masked diagonal tail.
  for (int j0 = 0; j0 < q0; j0 += 64)          process_block(j0, false);
  for (int j0 = q0; j0 < q0 + 128; j0 += 64)   process_block(j0, true);

  // ---- normalize and write [B*L, D] f16 for the output projection ----
#pragma unroll
  for (int r = 0; r < 8; r++) {
    int qi = q0 + wave * 16 + r + 8 * hf;
    float inv = 1.0f / lsum[r];
    size_t rowbase = ((size_t)b * LQ + qi) * DMODEL + h * DKH;
#pragma unroll
    for (int d = 0; d < 4; d++)
      O[rowbase + d * 16 + ln] = (_Float16)(oacc[d][r] * inv);
  }
}

// ---------------------------------------------------------------------------
// Launch
// ---------------------------------------------------------------------------
extern "C" void kernel_launch(void* const* d_in, const int* in_sizes, int n_in,
                              void* d_out, int out_size, void* d_ws,
                              size_t ws_size, hipStream_t stream) {
  (void)in_sizes; (void)n_in; (void)out_size; (void)ws_size;
  const float* x    = (const float*)d_in[0];
  // d_in[1] is the causal mask; applied analytically.
  const float* Wqkv = (const float*)d_in[2];
  const float* bqkv = (const float*)d_in[3];
  const float* Wout = (const float*)d_in[4];
  const float* bout = (const float*)d_in[5];
  float* out = (float*)d_out;

  char* ws = (char*)d_ws;
  _Float16* x16   = (_Float16*)(ws);                         // 16.78 MB
  _Float16* wqkvt = (_Float16*)(ws + (size_t)16777216);      //  6.29 MB
  _Float16* woutt = (_Float16*)(ws + (size_t)23068672);      //  2.10 MB
  _Float16* q16   = (_Float16*)(ws + (size_t)25165824);      // 16.78 MB
  _Float16* k16   = (_Float16*)(ws + (size_t)41943040);      // 16.78 MB
  _Float16* vv16  = (_Float16*)(ws + (size_t)58720256);      // 16.78 MB
  _Float16* a16   = (_Float16*)(ws + (size_t)75497472);      // 16.78 MB

  const int nx = BATCH * LQ * DMODEL;  // 8388608
  cvt_f32_f16<<<nx / 256, 256, 0, stream>>>(x, x16, nx);
  transpose_cvt<<<dim3(3 * DMODEL / 32, DMODEL / 8), 256, 0, stream>>>(
      Wqkv, wqkvt, DMODEL, 3 * DMODEL);
  transpose_cvt<<<dim3(DMODEL / 32, DMODEL / 8), 256, 0, stream>>>(
      Wout, woutt, DMODEL, DMODEL);

  // QKV projection: [8192,1024] x [1024,3072]
  gemm128<0><<<dim3(3 * DMODEL / 128, BATCH * LQ / 128), 256, 0, stream>>>(
      x16, wqkvt, bqkv, q16, k16, vv16, nullptr);

  // Flash attention over (q-block, b*h)
  flash_attn<<<dim3(LQ / 128, BATCH * NH), 256, 0, stream>>>(q16, k16, vv16, a16);

  // Output projection: [8192,1024] x [1024,1024] -> f32 d_out
  gemm128<1><<<dim3(DMODEL / 128, BATCH * LQ / 128), 256, 0, stream>>>(
      a16, woutt, bout, nullptr, nullptr, nullptr, out);
}
